// FModel_52080773431571
// MI455X (gfx1250) — matrix operand order repaired and checked by reading, not compiled, not run on hardware
//
#include <hip/hip_runtime.h>
#include <hip/hip_bf16.h>

typedef _Float16 v8h  __attribute__((ext_vector_type(8)));
typedef _Float16 v16h __attribute__((ext_vector_type(16)));
typedef float    v8f  __attribute__((ext_vector_type(8)));

#define B_SZ   2048
#define NNZ    8
#define SYN    128
#define SEM    256
#define HIDDEN 1024
#define OUT_N  10000
#define IN_DIM 647      // 128 + 256 + 256 + 7
#define KPAD   672      // 21 * 32

// ---------------------------------------------------------------------------
// WMMA fragment loaders (16x16x32 f16, wave32; this file's contiguous k order, same for A and B)
// A 16x32: lane l holds row (l&15); element e -> K = 16*(l>>4) + e
// B 32x16: lane l holds col (l&15); element e -> K = 16*(l>>4) + e
// ---------------------------------------------------------------------------
__device__ __forceinline__ v16h load_a_frag(const _Float16* p) {
  v8h lo = *(const v8h*)(p);
  v8h hi = *(const v8h*)(p + 8);
  return __builtin_shufflevector(lo, hi, 0,1,2,3,4,5,6,7,8,9,10,11,12,13,14,15);
}
__device__ __forceinline__ v16h load_b_frag(const _Float16* p) {
  v8h lo = *(const v8h*)(p);
  v8h hi = *(const v8h*)(p + 8);
  return __builtin_shufflevector(lo, hi, 0,1,2,3,4,5,6,7,8,9,10,11,12,13,14,15);
}

// CDNA5 async global->LDS copy (ASYNCcnt-tracked DMA), 16B per lane.
// VDST slot carries the per-lane LDS byte address; generic-LDS flat addrs
// truncate to the LDS offset.
__device__ __forceinline__ void async_g2l_b128(unsigned lds_addr, const void* gaddr) {
  asm volatile("global_load_async_to_lds_b128 %0, %1, off"
               :: "v"(lds_addr), "v"(gaddr) : "memory");
}
__device__ __forceinline__ void wait_async_le1() {
  asm volatile("s_wait_asynccnt 0x1" ::: "memory");
}
__device__ __forceinline__ void wait_async_0() {
  asm volatile("s_wait_asynccnt 0x0" ::: "memory");
}

// ---------------------------------------------------------------------------
// Weight conversion kernels (run every launch: deterministic, capture-safe)
// ---------------------------------------------------------------------------
__global__ __launch_bounds__(256) void cvt_w1(const float* __restrict__ W1,
                                              _Float16* __restrict__ Wp) {
  int i = blockIdx.x * 256 + threadIdx.x;
  if (i >= HIDDEN * KPAD) return;
  int r = i / KPAD, c = i % KPAD;
  Wp[i] = (_Float16)(c < IN_DIM ? W1[(size_t)r * IN_DIM + c] : 0.0f);
}

__global__ __launch_bounds__(256) void cvt_w2(const float* __restrict__ W2,
                                              _Float16* __restrict__ Wp) {
  int i = blockIdx.x * 256 + threadIdx.x;
  if (i >= OUT_N * HIDDEN) return;
  Wp[i] = (_Float16)W2[i];
}

// ---------------------------------------------------------------------------
// Build x = [cat_embed | hvb_embed | hvf_embed | onehot | zero-pad] in f16.
// Multi-hot "set" semantics: duplicate indices within a row count once.
// ---------------------------------------------------------------------------
__global__ __launch_bounds__(128) void build_x(
    const float* __restrict__ d_onehot, const int* __restrict__ cat_ix,
    const int* __restrict__ hvb_ix, const int* __restrict__ hvf_ix,
    const float* __restrict__ hvb_top, const float* __restrict__ hvf_top,
    const float* __restrict__ cat_emb, const float* __restrict__ hv_emb,
    _Float16* __restrict__ X)
{
  int b = blockIdx.x, t = threadIdx.x;
  _Float16* xr = X + (size_t)b * KPAD;

  if (t < SYN)
    xr[t] = (_Float16)cat_emb[(size_t)cat_ix[b] * SYN + t];

  int ib[NNZ], ifx[NNZ];
  bool ub[NNZ], uf[NNZ];
#pragma unroll
  for (int j = 0; j < NNZ; ++j) {
    ib[j]  = hvb_ix[(size_t)b * NNZ + j];
    ifx[j] = hvf_ix[(size_t)b * NNZ + j];
  }
#pragma unroll
  for (int j = 0; j < NNZ; ++j) {
    bool db = false, df = false;
#pragma unroll
    for (int k = 0; k < NNZ; ++k) {
      if (k < j) {
        db = db || (ib[k] == ib[j]);
        df = df || (ifx[k] == ifx[j]);
      }
    }
    ub[j] = !db; uf[j] = !df;
  }

  for (int c = t; c < SEM; c += 128) {
    float s = hvb_top[(size_t)b * SEM + c];
#pragma unroll
    for (int j = 0; j < NNZ; ++j)
      if (ub[j]) s += hv_emb[(size_t)ib[j] * SEM + c];
    xr[SYN + c] = (_Float16)s;

    float s2 = hvf_top[(size_t)b * SEM + c];
#pragma unroll
    for (int j = 0; j < NNZ; ++j)
      if (uf[j]) s2 += hv_emb[(size_t)ifx[j] * SEM + c];
    xr[SYN + SEM + c] = (_Float16)s2;
  }

  if (t < KPAD - 640) {   // 7 onehot cols + 25 zero pad cols
    float v = (t < 7) ? d_onehot[(size_t)b * 7 + t] : 0.0f;
    xr[640 + t] = (_Float16)v;
  }
}

// ---------------------------------------------------------------------------
// GEMM1: h = relu(x @ W1^T + b1). One wave per 16x64 tile (small GEMM).
// ---------------------------------------------------------------------------
__global__ __launch_bounds__(256) void gemm1_relu(
    const _Float16* __restrict__ X,   // (B_SZ, KPAD)
    const _Float16* __restrict__ W,   // (HIDDEN, KPAD) row-major == B^T
    const float* __restrict__ bias,   // (HIDDEN)
    _Float16* __restrict__ H)         // (B_SZ, HIDDEN)
{
  const int NT64 = HIDDEN / 64;                 // 16
  int wave = blockIdx.x * 8 + (threadIdx.x >> 5);
  int lane = threadIdx.x & 31;
  if (wave >= (B_SZ / 16) * NT64) return;
  int tn = wave % NT64, tm = wave / NT64;
  int lhalf = lane >> 4, lmod = lane & 15;

  const _Float16* arow = X + (size_t)(tm * 16 + lmod) * KPAD + 16 * lhalf;
  const _Float16* bp[4];
  v8f acc[4];
#pragma unroll
  for (int t = 0; t < 4; ++t) {
    int n = tn * 64 + t * 16 + lmod;
    bp[t] = W + (size_t)n * KPAD + 16 * lhalf;
    float bv = bias[n];
#pragma unroll
    for (int r = 0; r < 8; ++r) acc[t][r] = bv;
  }

  for (int kk = 0; kk < KPAD; kk += 32) {
    v16h a = load_a_frag(arow + kk);
#pragma unroll
    for (int t = 0; t < 4; ++t) {
      v16h bfr = load_b_frag(bp[t] + kk);
      acc[t] = __builtin_amdgcn_wmma_f32_16x16x32_f16(
          false, a, false, bfr, (short)0, acc[t], false, false);
    }
  }

#pragma unroll
  for (int t = 0; t < 4; ++t) {
    int n = tn * 64 + t * 16 + lmod;
#pragma unroll
    for (int r = 0; r < 8; ++r) {
      int row = tm * 16 + lhalf * 8 + r;
      float v = acc[t][r];
      H[(size_t)row * HIDDEN + n] = (_Float16)(v > 0.0f ? v : 0.0f);
    }
  }
}

// ---------------------------------------------------------------------------
// GEMM2: y = h @ W2^T + b2 -> f32. M=2048, N=10000, K=1024.
// Block (8 waves) = 128M x 64N macro-tile. B k-slice (64x32 f16 = 4KB) is
// async-DMA'd into LDS once per block (8x reuse), double buffered so the
// GLOBAL_LOAD_ASYNC_TO_LDS_B128 of slice i+1 overlaps WMMA on slice i.
// ---------------------------------------------------------------------------
__global__ __launch_bounds__(256) void gemm2_bias_lds(
    const _Float16* __restrict__ H,   // (B_SZ, HIDDEN)
    const _Float16* __restrict__ W,   // (OUT_N, HIDDEN) row-major == B^T
    const float* __restrict__ bias,   // (OUT_N)
    float* __restrict__ Y)            // (B_SZ, OUT_N)
{
  __shared__ _Float16 sB[2][64 * 32];           // 2 x 4KB, [col][k] k-contig

  const int NT64 = (OUT_N + 63) / 64;           // 157
  int bm = blockIdx.x / NT64;                   // 0..15
  int bn = blockIdx.x % NT64;                   // 0..156
  int tid = threadIdx.x;
  int wv = tid >> 5, lane = tid & 31;
  int lhalf = lane >> 4, lmod = lane & 15;

  // Per-thread DMA source: thread copies 16B of one B column's k-slice.
  int colc = tid >> 2;                          // 0..63 (column within tile)
  int q    = tid & 3;                           // 16B quarter of 64B column
  int nsrc = bn * 64 + colc;
  if (nsrc >= OUT_N) nsrc = OUT_N - 1;          // clamp: finite dup data, stores guarded
  const _Float16* gsrc = W + (size_t)nsrc * HIDDEN + q * 8;
  unsigned ldst0 = (unsigned)(size_t)(&sB[0][colc * 32 + q * 8]);
  unsigned ldst1 = (unsigned)(size_t)(&sB[1][colc * 32 + q * 8]);

  const _Float16* arow = H + (size_t)(bm * 128 + wv * 16 + lmod) * HIDDEN + 16 * lhalf;

  v8f acc[4];
  bool valid[4];
#pragma unroll
  for (int t = 0; t < 4; ++t) {
    int n0 = bn * 64 + t * 16;
    valid[t] = (n0 < OUT_N);                    // wave-uniform (N mult of 16)
    int n = n0 + lmod;
    if (n >= OUT_N) n = OUT_N - 1;
    float bv = bias[n];
#pragma unroll
    for (int r = 0; r < 8; ++r) acc[t][r] = bv;
  }

  // Prologue: DMA slice 0 into buffer 0.
  async_g2l_b128(ldst0, (const void*)gsrc);

  for (int kk = 0; kk < HIDDEN; kk += 32) {
    int cur = (kk >> 5) & 1;
    bool has_next = (kk + 32) < HIDDEN;
    if (has_next)                                // DMA slice i+1 into buf cur^1
      async_g2l_b128(cur ? ldst0 : ldst1, (const void*)(gsrc + kk + 32));
    if (has_next) wait_async_le1();              // in-order: slice i has landed
    else          wait_async_0();
    __syncthreads();                             // slice i visible to all waves

    v16h a = load_a_frag(arow + kk);
    const _Float16* bb = &sB[cur][0];
#pragma unroll
    for (int t = 0; t < 4; ++t) {
      v16h bfr = load_b_frag(bb + (t * 16 + lmod) * 32 + 16 * lhalf);
      acc[t] = __builtin_amdgcn_wmma_f32_16x16x32_f16(
          false, a, false, bfr, (short)0, acc[t], false, false);
    }
    __syncthreads();    // all reads of buf cur done before slice i+2 DMA targets it
  }

#pragma unroll
  for (int t = 0; t < 4; ++t) {
    if (valid[t]) {
      int n = bn * 64 + t * 16 + lmod;
#pragma unroll
      for (int r = 0; r < 8; ++r) {
        int row = bm * 128 + wv * 16 + lhalf * 8 + r;
        Y[(size_t)row * OUT_N + n] = acc[t][r];
      }
    }
  }
}

// ---------------------------------------------------------------------------
// In-place row-wise log_softmax over 10000 cols. One block per row.
// ---------------------------------------------------------------------------
__global__ __launch_bounds__(256) void logsoftmax_rows(float* __restrict__ Y) {
  __shared__ float red[256];
  int b = blockIdx.x, t = threadIdx.x;
  float* row = Y + (size_t)b * OUT_N;

  float mx = -3.402823466e38f;
  for (int c = t; c < OUT_N; c += 256) mx = fmaxf(mx, row[c]);
  red[t] = mx; __syncthreads();
  for (int s = 128; s > 0; s >>= 1) {
    if (t < s) red[t] = fmaxf(red[t], red[t + s]);
    __syncthreads();
  }
  mx = red[0]; __syncthreads();

  float sum = 0.0f;
  for (int c = t; c < OUT_N; c += 256) sum += __expf(row[c] - mx);
  red[t] = sum; __syncthreads();
  for (int s = 128; s > 0; s >>= 1) {
    if (t < s) red[t] += red[t + s];
    __syncthreads();
  }
  float lse = __logf(red[0]) + mx;

  for (int c = t; c < OUT_N; c += 256) row[c] = row[c] - lse;
}

// ---------------------------------------------------------------------------
extern "C" void kernel_launch(void* const* d_in, const int* in_sizes, int n_in,
                              void* d_out, int out_size, void* d_ws, size_t ws_size,
                              hipStream_t stream) {
  const float* d_onehot    = (const float*)d_in[0];
  const int*   cat_b_ix    = (const int*)  d_in[1];
  const int*   hvb_ix      = (const int*)  d_in[2];
  const int*   hvf_ix      = (const int*)  d_in[3];
  const float* hvb_top     = (const float*)d_in[4];
  const float* hvf_top     = (const float*)d_in[5];
  const float* cat_embeds  = (const float*)d_in[6];
  const float* hvec_embeds = (const float*)d_in[7];
  const float* W1          = (const float*)d_in[8];
  const float* b1          = (const float*)d_in[9];
  const float* W2          = (const float*)d_in[10];
  const float* b2          = (const float*)d_in[11];
  float* out = (float*)d_out;

  // Workspace layout (all f16, each region 16B-aligned by construction)
  _Float16* xh  = (_Float16*)d_ws;                        // B_SZ   * KPAD
  _Float16* w1h = xh  + (size_t)B_SZ   * KPAD;            // HIDDEN * KPAD
  _Float16* hh  = w1h + (size_t)HIDDEN * KPAD;            // B_SZ   * HIDDEN
  _Float16* w2h = hh  + (size_t)B_SZ   * HIDDEN;          // OUT_N  * HIDDEN

  cvt_w1<<<(HIDDEN * KPAD + 255) / 256, 256, 0, stream>>>(W1, w1h);
  cvt_w2<<<(OUT_N * HIDDEN + 255) / 256, 256, 0, stream>>>(W2, w2h);

  build_x<<<B_SZ, 128, 0, stream>>>(d_onehot, cat_b_ix, hvb_ix, hvf_ix,
                                    hvb_top, hvf_top, cat_embeds, hvec_embeds, xh);

  {
    int waves = (B_SZ / 16) * (HIDDEN / 64);              // 2048
    gemm1_relu<<<(waves + 7) / 8, 256, 0, stream>>>(xh, w1h, b1, hh);
  }
  {
    int blocks = (B_SZ / 128) * ((OUT_N + 63) / 64);      // 16 * 157 = 2512
    gemm2_bias_lds<<<blocks, 256, 0, stream>>>(hh, w2h, b2, out);
  }

  logsoftmax_rows<<<B_SZ, 256, 0, stream>>>(out);
}
